// HeteroGAT_87299505258972
// MI455X (gfx1250) — matrix operand order repaired and checked by reading, not compiled, not run on hardware
//
#include <hip/hip_runtime.h>

// ---------------- problem constants ----------------
#define CH    128         // HEADS*H
#define HD    64
#define NEGS  0.2f
#define BNEPS 1e-5f
#define NR    40000
#define NV    4000
#define E_RR  400000
#define E_VR  150000
#define E_RV  150000
#define FIN   16
#define EATTR 8

typedef __attribute__((ext_vector_type(16))) _Float16 v16h;
typedef __attribute__((ext_vector_type(8)))  _Float16 v8h;
typedef __attribute__((ext_vector_type(8)))  float    v8f;

__device__ __forceinline__ float fixnan(float v) { return __builtin_isnan(v) ? 0.f : v; }
__device__ __forceinline__ float lrelu(float v)  { return v >= 0.f ? v : NEGS * v; }

// order-preserving float<->uint mapping for atomic max on signed floats
__device__ __forceinline__ unsigned fkey(float f) {
  unsigned u = __float_as_uint(f);
  return (u & 0x80000000u) ? ~u : (u | 0x80000000u);
}
__device__ __forceinline__ float kval(unsigned k) {
  unsigned u = (k & 0x80000000u) ? (k & 0x7FFFFFFFu) : ~k;
  return __uint_as_float(u);
}

// gfx1250 async global->LDS copy (ASYNCcnt-tracked), per-lane LDS destinations
__device__ __forceinline__ void async_b128(unsigned ldsoff, const void* g) {
  asm volatile("global_load_async_to_lds_b128 %0, %1, off"
               :: "v"(ldsoff), "v"(g) : "memory");
}
__device__ __forceinline__ void wait_async0() {
  asm volatile("s_wait_asynccnt 0x0" ::: "memory");
}

// ---------------- WMMA GEMM: C[M,128] = fixnan(A[M,K]) @ W[K,128] ----------------
// 256 threads = 8 wave32; block tile 128 rows x 128 cols; K staged in 32-slabs.
// A staged f32 (async global->LDS for full interior tiles), W staged f16 TRANSPOSED
// so B fragments are contiguous ds_load_b128 pairs.
__global__ __launch_bounds__(256)
void gemm_wmma(const float* __restrict__ A, const float* __restrict__ W,
               float* __restrict__ C, int M, int K)
{
  __shared__ float    Af[128][36];   // 18 KB, 144B row stride (16B-mult, bank-skewed)
  __shared__ _Float16 Bst[CH][56];   // 14 KB, 112B row stride (16B-mult, bank-skewed)
  const int tid  = threadIdx.x;
  const int wave = tid >> 5;
  const int lane = tid & 31;
  const int rowBase = blockIdx.x * 128;
  const bool fullM = (rowBase + 128 <= M);
  const unsigned ldsA = (unsigned)(size_t)(&Af[0][0]);

  v8f acc[8];
  #pragma unroll
  for (int t = 0; t < 8; ++t)
    #pragma unroll
    for (int i = 0; i < 8; ++i) acc[t][i] = 0.f;

  for (int k0 = 0; k0 < K; k0 += 32) {
    const bool use_async = fullM && (k0 + 32 <= K);
    if (use_async) {
      // each wave DMAs 16 rows x 32 f32: 4 async b128 ops, 4 rows x 128B per op
      const int r0  = (wave << 4) + (lane >> 3);
      const int col = (lane & 7) * 4;
      #pragma unroll
      for (int c = 0; c < 4; ++c) {
        const int row = r0 + 4 * c;
        async_b128(ldsA + (unsigned)((row * 36 + col) * 4),
                   A + (size_t)(rowBase + row) * K + (k0 + col));
      }
    } else {
      for (int i = tid; i < 128 * 32; i += 256) {
        int r = i >> 5, k = i & 31;
        int gr = rowBase + r, gk = k0 + k;
        float v = 0.f;
        if (gr < M && gk < K) v = fixnan(A[(size_t)gr * K + gk]);
        Af[r][k] = v;
      }
    }
    // stage W tile transposed: Bst[n][k] (coalesced global reads)
    for (int i = tid; i < 32 * CH; i += 256) {
      int k = i >> 7, n = i & (CH - 1);
      int gk = k0 + k;
      float v = (gk < K) ? W[(size_t)gk * CH + n] : 0.f;
      Bst[n][k] = (_Float16)v;
    }
    if (use_async) wait_async0();
    __syncthreads();

    // A fragment: lanes<16: row M, K0-7 & K16-23; lanes>=16: K8-15 & K24-31
    const int ar  = (wave << 4) + (lane & 15);
    const int akb = (lane < 16) ? 0 : 8;
    const float4 a0 = *(const float4*)&Af[ar][akb];
    const float4 a1 = *(const float4*)&Af[ar][akb + 4];
    const float4 a2 = *(const float4*)&Af[ar][akb + 16];
    const float4 a3 = *(const float4*)&Af[ar][akb + 20];
    v16h af;
    af[0]  = (_Float16)a0.x; af[1]  = (_Float16)a0.y; af[2]  = (_Float16)a0.z; af[3]  = (_Float16)a0.w;
    af[4]  = (_Float16)a1.x; af[5]  = (_Float16)a1.y; af[6]  = (_Float16)a1.z; af[7]  = (_Float16)a1.w;
    af[8]  = (_Float16)a2.x; af[9]  = (_Float16)a2.y; af[10] = (_Float16)a2.z; af[11] = (_Float16)a2.w;
    af[12] = (_Float16)a3.x; af[13] = (_Float16)a3.y; af[14] = (_Float16)a3.z; af[15] = (_Float16)a3.w;

    const int bkb = (lane < 16) ? 0 : 8;
    #pragma unroll
    for (int nt = 0; nt < 8; ++nt) {
      // B fragment: K=kb..kb+7 then kb+16..kb+23 for column n (matches A) -> two contiguous runs in Bst[n][*]
      const int bn = (nt << 4) + (lane & 15);
      const v8h b0 = *(const v8h*)&Bst[bn][bkb];
      const v8h b1 = *(const v8h*)&Bst[bn][bkb + 16];
      const v16h bf = __builtin_shufflevector(b0, b1, 0, 1, 2, 3, 4, 5, 6, 7,
                                              8, 9, 10, 11, 12, 13, 14, 15);
      acc[nt] = __builtin_amdgcn_wmma_f32_16x16x32_f16(
          false, af, false, bf, (short)0, acc[nt], false, false);
    }
    __syncthreads();
  }

  // store C (lanes<16 -> M=r, lanes>=16 -> M=8+r; N = lane&15 within 16-col tile)
  const int n0   = lane & 15;
  const int rsel = (lane < 16) ? 0 : 8;
  #pragma unroll
  for (int nt = 0; nt < 8; ++nt) {
    const int col = (nt << 4) + n0;
    #pragma unroll
    for (int r = 0; r < 8; ++r) {
      const int row = rowBase + (wave << 4) + rsel + r;
      if (row < M) C[(size_t)row * CH + col] = acc[nt][r];
    }
  }
}

// ---------------- fold attention vector into weight: weff[k,h] = sum_j W[k, h*64+j]*a[h,j] ----------------
__global__ void prep_weff(const float* __restrict__ W, const float* __restrict__ a,
                          float* __restrict__ weff, int K)
{
  int i = blockIdx.x * blockDim.x + threadIdx.x;
  if (i >= K * 2) return;
  int k = i >> 1, h = i & 1;
  float s = 0.f;
  #pragma unroll 8
  for (int j = 0; j < HD; ++j) s += W[(size_t)k * CH + h * HD + j] * a[h * HD + j];
  weff[i] = s;
}

// ---------------- attention scores: out[i,h] = fixnan(x[i,:]) . weff[:,h] ----------------
__global__ void attn_score(const float* __restrict__ x, const float* __restrict__ weff,
                           float* __restrict__ out, int N, int K)
{
  int i = blockIdx.x * blockDim.x + threadIdx.x;
  if (i >= N) return;
  float s0 = 0.f, s1 = 0.f;
  for (int k = 0; k < K; ++k) {
    float v = fixnan(x[(size_t)i * K + k]);
    s0 += v * weff[2 * k];
    s1 += v * weff[2 * k + 1];
  }
  out[2 * i] = s0; out[2 * i + 1] = s1;
}

// ---------------- init helpers ----------------
__global__ void fill_bias(float* __restrict__ o, const float* __restrict__ b0,
                          const float* __restrict__ b1, float w, int n)
{
  int i = blockIdx.x * blockDim.x + threadIdx.x;
  if (i < n) { int c = i & (CH - 1); o[i] = w * (b0[c] + b1[c]); }
}
__global__ void zero_f(float* p, int n)
{
  int i = blockIdx.x * blockDim.x + threadIdx.x;
  if (i < n) p[i] = 0.f;
}
__global__ void init_mden(unsigned* mk, float* dn, int n)
{
  int i = blockIdx.x * blockDim.x + threadIdx.x;
  if (i < n) { mk[i] = 0x007FFFFFu; dn[i] = 0.f; }  // fkey(-inf)
}

// ---------------- edge pass 1: e = lrelu(es[src]+ed[dst]); segment max into mkey ----------------
__global__ void epass1(const int* __restrict__ src, const int* __restrict__ dst,
                       const float* __restrict__ es, const float* __restrict__ ed,
                       float* __restrict__ ee, unsigned* __restrict__ mkey, int E)
{
  int e = blockIdx.x * blockDim.x + threadIdx.x;
  if (e >= E) return;
  int s = src[e], d = dst[e];
  #pragma unroll
  for (int h = 0; h < 2; ++h) {
    float v = lrelu(es[2 * s + h] + ed[2 * d + h]);
    ee[2 * e + h] = v;
    atomicMax(&mkey[2 * d + h], fkey(v));
  }
}

// ---------------- edge pass 2: ex = exp(e - m[dst]); segment sum into den ----------------
__global__ void epass2(const int* __restrict__ dst, const float* __restrict__ ee,
                       const unsigned* __restrict__ mkey, float* __restrict__ den,
                       float* __restrict__ exe, int E)
{
  int e = blockIdx.x * blockDim.x + threadIdx.x;
  if (e >= E) return;
  int d = dst[e];
  #pragma unroll
  for (int h = 0; h < 2; ++h) {
    float m = kval(mkey[2 * d + h]);
    float x = __expf(ee[2 * e + h] - m);
    exe[2 * e + h] = x;
    atomicAdd(&den[2 * d + h], x);
  }
}

// ---------------- edge pass 3: out[dst] += hs[src] * alpha (one wave32 per edge, 4 ch/lane) ----------------
__global__ __launch_bounds__(256)
void epass3(const int* __restrict__ src, const int* __restrict__ dst,
            const float* __restrict__ hs, const float* __restrict__ exe,
            const float* __restrict__ den, float* __restrict__ out,
            int E, float scale)
{
  int gid  = blockIdx.x * blockDim.x + threadIdx.x;
  int e    = gid >> 5;
  int lane = gid & 31;
  if (e >= E) return;
  int s = src[e], d = dst[e];

  // prefetch a lookahead edge's hs row into cache
  int e2 = e + 64;
  if (lane == 0 && e2 < E) __builtin_prefetch(&hs[(size_t)src[e2] * CH], 0, 0);

  int c = lane * 4;              // 0..124, never crosses head boundary (64 % 4 == 0)
  int h = c >> 6;
  float alpha = scale * exe[2 * e + h] / den[2 * d + h];
  const float4 hv = *(const float4*)(hs + (size_t)s * CH + c);
  float* o = out + (size_t)d * CH + c;
  atomicAdd(o + 0, hv.x * alpha);
  atomicAdd(o + 1, hv.y * alpha);
  atomicAdd(o + 2, hv.z * alpha);
  atomicAdd(o + 3, hv.w * alpha);
}

// ---------------- per-column sum / sumsq (coalesced grid-stride, atomic combine) ----------------
__global__ __launch_bounds__(256)
void colsum(const float* __restrict__ x, int N, float* __restrict__ sum, float* __restrict__ sumsq)
{
  int c     = threadIdx.x & (CH - 1);
  int rhalf = threadIdx.x >> 7;   // 0 or 1
  float s = 0.f, s2 = 0.f;
  for (int r = blockIdx.x * 2 + rhalf; r < N; r += gridDim.x * 2) {
    float v = x[(size_t)r * CH + c];
    s += v; s2 += v * v;
  }
  __shared__ float sh[256];
  sh[threadIdx.x] = s; __syncthreads();
  if (rhalf == 0) atomicAdd(&sum[c], sh[c] + sh[128 + c]);
  __syncthreads();
  sh[threadIdx.x] = s2; __syncthreads();
  if (rhalf == 0) atomicAdd(&sumsq[c], sh[c] + sh[128 + c]);
}

// ---------------- BN (training-mode biased stats) + leaky relu ----------------
__global__ void bn_apply(const float* __restrict__ o, float* __restrict__ xo,
                         const float* __restrict__ stats, const float* __restrict__ g,
                         const float* __restrict__ bt, int n, float invN)
{
  int i = blockIdx.x * blockDim.x + threadIdx.x;
  if (i >= n) return;
  int c = i & (CH - 1);
  float mu  = stats[c] * invN;
  float var = stats[CH + c] * invN - mu * mu;
  float v = (o[i] - mu) * rsqrtf(var + BNEPS) * g[c] + bt[c];
  xo[i] = lrelu(v);
}

// ---------------- node projections for output head: p[i,:] = x[i,:] @ Wl(128x2) ----------------
__global__ void node_proj(const float* __restrict__ x, const float* __restrict__ Wl,
                          float* __restrict__ p, int N)
{
  int i = blockIdx.x * blockDim.x + threadIdx.x;
  if (i >= N) return;
  float s0 = 0.f, s1 = 0.f;
  for (int c = 0; c < CH; ++c) {
    float v = x[(size_t)i * CH + c];
    s0 += v * Wl[2 * c];
    s1 += v * Wl[2 * c + 1];
  }
  p[2 * i] = s0; p[2 * i + 1] = s1;
}

// ---------------- fold edge-attr head: wp2[rel][k,o] = Wp[rel] @ Wlin[128:192]; cst = bp@... + blin ----------------
__global__ void prep_wp2(const float* __restrict__ Wp, const float* __restrict__ bp,
                         const float* __restrict__ Wlin, const float* __restrict__ blin,
                         float* __restrict__ out)
{
  int t = threadIdx.x;
  if (t < 48) {
    int rel = t >> 4, idx = t & 15, k = idx >> 1, o = idx & 1;
    float s = 0.f;
    for (int h = 0; h < HD; ++h) s += Wp[(size_t)rel * EATTR * HD + k * HD + h] * Wlin[(128 + h) * 2 + o];
    out[rel * 18 + idx] = s;
  } else if (t < 54) {
    int u = t - 48, rel = u >> 1, o = u & 1;
    float s = blin[o];
    for (int h = 0; h < HD; ++h) s += bp[rel * HD + h] * Wlin[(128 + h) * 2 + o];
    out[rel * 18 + 16 + o] = s;
  }
}

// ---------------- output head: per-edge logits + 2-way softmax ----------------
__global__ void edge_out(const int* __restrict__ src, const int* __restrict__ dst,
                         const float* __restrict__ ea, const float* __restrict__ pS,
                         const float* __restrict__ pD, const float* __restrict__ wc,
                         float* __restrict__ out, int E)
{
  int e = blockIdx.x * blockDim.x + threadIdx.x;
  if (e >= E) return;
  int s = src[e], d = dst[e];
  float l0 = pS[2 * s]     + pD[2 * d]     + wc[16];
  float l1 = pS[2 * s + 1] + pD[2 * d + 1] + wc[17];
  #pragma unroll
  for (int k = 0; k < EATTR; ++k) {
    float v = fixnan(ea[(size_t)e * EATTR + k]);
    l0 += v * wc[2 * k];
    l1 += v * wc[2 * k + 1];
  }
  float m  = fmaxf(l0, l1);
  float e0 = __expf(l0 - m), e1 = __expf(l1 - m);
  float inv = 1.f / (e0 + e1);
  out[2 * e] = e0 * inv; out[2 * e + 1] = e1 * inv;
}

// ==================================================================
extern "C" void kernel_launch(void* const* d_in, const int* in_sizes, int n_in,
                              void* d_out, int out_size, void* d_ws, size_t ws_size,
                              hipStream_t stream)
{
  const float* x_req  = (const float*)d_in[0];
  const float* x_veh  = (const float*)d_in[1];
  const int*   ei_rr  = (const int*)d_in[2];
  const int*   ei_vr  = (const int*)d_in[3];
  const int*   ei_rv  = (const int*)d_in[4];
  const float* ea_rr  = (const float*)d_in[5];
  const float* ea_vr  = (const float*)d_in[6];
  const float* ea_rv  = (const float*)d_in[7];
  const float* Wsrc1  = (const float*)d_in[8];
  const float* Wdst1  = (const float*)d_in[9];
  const float* asrc1  = (const float*)d_in[10];
  const float* adst1  = (const float*)d_in[11];
  const float* bias1  = (const float*)d_in[12];
  const float* Wsrc2  = (const float*)d_in[13];
  const float* Wdst2  = (const float*)d_in[14];
  const float* asrc2  = (const float*)d_in[15];
  const float* adst2  = (const float*)d_in[16];
  const float* bias2  = (const float*)d_in[17];
  const float* bn_g   = (const float*)d_in[18];
  const float* bn_b   = (const float*)d_in[19];
  const float* Wp     = (const float*)d_in[20];
  const float* bp     = (const float*)d_in[21];
  const float* Wlin   = (const float*)d_in[22];
  const float* blin   = (const float*)d_in[23];
  float* out = (float*)d_out;

  // ---- workspace carve-up (~74 MB of f32) ----
  float* Wk = (float*)d_ws;
  size_t off = 0;
  auto alloc = [&](size_t n) { float* p = Wk + off; off += n; return p; };
  float*    xr    = alloc((size_t)NR * CH);
  float*    xv    = alloc((size_t)NV * CH);
  float*    hs    = alloc((size_t)NR * CH);
  float*    o_req = alloc((size_t)NR * CH);
  float*    o_veh = alloc((size_t)NV * CH);
  float*    es    = alloc((size_t)NR * 2);
  float*    ed    = alloc((size_t)NR * 2);
  float*    ee    = alloc((size_t)E_RR * 2);
  float*    exe   = alloc((size_t)E_RR * 2);
  unsigned* mkey  = (unsigned*)alloc((size_t)NR * 2);
  float*    den   = alloc((size_t)NR * 2);
  float*    stats = alloc(CH * 2);
  float*    weffS = alloc(CH * 2);
  float*    weffD = alloc(CH * 2);
  float*    pSr   = alloc((size_t)NR * 2);
  float*    pDr   = alloc((size_t)NR * 2);
  float*    pSv   = alloc((size_t)NV * 2);
  float*    pDv   = alloc((size_t)NV * 2);
  float*    wp2   = alloc(3 * 18);

  auto cdiv = [](long long a, long long b) { return (unsigned)((a + b - 1) / b); };

  for (int l = 0; l < 2; ++l) {
    const float* Xr = (l == 0) ? x_req : xr;
    const float* Xv = (l == 0) ? x_veh : xv;
    const int K = (l == 0) ? FIN : CH;
    const float* Ws_l = (l == 0) ? Wsrc1 : Wsrc2;
    const float* Wd_l = (l == 0) ? Wdst1 : Wdst2;
    const float* as_l = (l == 0) ? asrc1 : asrc2;
    const float* ad_l = (l == 0) ? adst1 : adst2;
    const float* b_l  = (l == 0) ? bias1 : bias2;
    const size_t wstr = (size_t)K * CH;

    // init aggregation targets with (averaged) biases
    fill_bias<<<cdiv((size_t)NR * CH, 256), 256, 0, stream>>>(o_req, b_l + 0 * CH, b_l + 1 * CH, 0.5f, NR * CH);
    fill_bias<<<cdiv((size_t)NV * CH, 256), 256, 0, stream>>>(o_veh, b_l + 2 * CH, b_l + 2 * CH, 0.5f, NV * CH);

    struct Rel { const int* ei; int E, Ns, Nd; const float* xs; const float* xd; float* ot; float sc; };
    Rel rels[3] = {
      { ei_rr, E_RR, NR, NR, Xr, Xr, o_req, 0.5f },
      { ei_vr, E_VR, NV, NR, Xv, Xr, o_req, 0.5f },
      { ei_rv, E_RV, NR, NV, Xr, Xv, o_veh, 1.0f },
    };

    for (int r = 0; r < 3; ++r) {
      const Rel& R = rels[r];
      const int* src = R.ei;
      const int* dst = R.ei + R.E;

      gemm_wmma<<<cdiv(R.Ns, 128), 256, 0, stream>>>(R.xs, Ws_l + r * wstr, hs, R.Ns, K);
      prep_weff<<<cdiv(K * 2, 256), 256, 0, stream>>>(Ws_l + r * wstr, as_l + r * CH, weffS, K);
      prep_weff<<<cdiv(K * 2, 256), 256, 0, stream>>>(Wd_l + r * wstr, ad_l + r * CH, weffD, K);
      attn_score<<<cdiv(R.Ns, 256), 256, 0, stream>>>(R.xs, weffS, es, R.Ns, K);
      attn_score<<<cdiv(R.Nd, 256), 256, 0, stream>>>(R.xd, weffD, ed, R.Nd, K);
      init_mden<<<cdiv(R.Nd * 2, 256), 256, 0, stream>>>(mkey, den, R.Nd * 2);
      epass1<<<cdiv(R.E, 256), 256, 0, stream>>>(src, dst, es, ed, ee, mkey, R.E);
      epass2<<<cdiv(R.E, 256), 256, 0, stream>>>(dst, ee, mkey, den, exe, R.E);
      epass3<<<cdiv((size_t)R.E * 32, 256), 256, 0, stream>>>(src, dst, hs, exe, den, R.ot, R.E, R.sc);
    }

    // BN + lrelu -> next-layer features
    zero_f<<<1, 256, 0, stream>>>(stats, CH * 2);
    colsum<<<512, 256, 0, stream>>>(o_req, NR, stats, stats + CH);
    bn_apply<<<cdiv((size_t)NR * CH, 256), 256, 0, stream>>>(
        o_req, xr, stats, bn_g + (l * 2 + 0) * CH, bn_b + (l * 2 + 0) * CH, NR * CH, 1.f / NR);

    zero_f<<<1, 256, 0, stream>>>(stats, CH * 2);
    colsum<<<128, 256, 0, stream>>>(o_veh, NV, stats, stats + CH);
    bn_apply<<<cdiv((size_t)NV * CH, 256), 256, 0, stream>>>(
        o_veh, xv, stats, bn_g + (l * 2 + 1) * CH, bn_b + (l * 2 + 1) * CH, NV * CH, 1.f / NV);
  }

  // ---- output head: per-node projections + folded edge-attr weights ----
  node_proj<<<cdiv(NR, 256), 256, 0, stream>>>(xr, Wlin,            pSr, NR);
  node_proj<<<cdiv(NR, 256), 256, 0, stream>>>(xr, Wlin + 192 * 2,  pDr, NR);
  node_proj<<<cdiv(NV, 256), 256, 0, stream>>>(xv, Wlin,            pSv, NV);
  node_proj<<<cdiv(NV, 256), 256, 0, stream>>>(xv, Wlin + 192 * 2,  pDv, NV);
  prep_wp2<<<1, 64, 0, stream>>>(Wp, bp, Wlin, blin, wp2);

  edge_out<<<cdiv(E_RR, 256), 256, 0, stream>>>(ei_rr, ei_rr + E_RR, ea_rr, pSr, pDr, wp2 + 0,  out, E_RR);
  edge_out<<<cdiv(E_VR, 256), 256, 0, stream>>>(ei_vr, ei_vr + E_VR, ea_vr, pSv, pDr, wp2 + 18, out + (size_t)E_RR * 2, E_VR);
  edge_out<<<cdiv(E_RV, 256), 256, 0, stream>>>(ei_rv, ei_rv + E_RV, ea_rv, pSr, pDv, wp2 + 36, out + (size_t)(E_RR + E_VR) * 2, E_RV);
}
